// PhysicsExpert_8151847928380
// MI455X (gfx1250) — matrix operand order repaired and checked by reading, not compiled, not run on hardware
//
#include <hip/hip_runtime.h>

// ---------------------------------------------------------------------------
// Types for CDNA5 WMMA (wave32, 16x16x32 bf16 -> f32)
// ---------------------------------------------------------------------------
typedef __attribute__((ext_vector_type(16))) __bf16        v16bf;
typedef __attribute__((ext_vector_type(8)))  float         v8f;
typedef __attribute__((ext_vector_type(8)))  unsigned int  v8u;

// f32 -> bf16 (round-to-nearest-even). Use the native bf16 type so the
// backend can emit the hardware cvt op on gfx1250 instead of a software
// bit-twiddle sequence.
__device__ __forceinline__ unsigned short f2bf(float f) {
  __bf16 h = (__bf16)f;
  return __builtin_bit_cast(unsigned short, h);
}

__device__ __forceinline__ v8f vzero8() {
  v8f z = {0.f, 0.f, 0.f, 0.f, 0.f, 0.f, 0.f, 0.f};
  return z;
}

__device__ __forceinline__ v16bf pack_frag(uint4 lo, uint4 hi) {
  v8u u;
  u[0] = lo.x; u[1] = lo.y; u[2] = lo.z; u[3] = lo.w;
  u[4] = hi.x; u[5] = hi.y; u[6] = hi.z; u[7] = hi.w;
  return __builtin_bit_cast(v16bf, u);
}

// B fragment (32x16 bf16). Wt is bf16 weights stored transposed: [Npad][Kpad]
// (K contiguous per output column). Contiguous k: VGPR v holds K=2v,2v+1 for
// lanes 0-15 (K+0) / lanes 16-31 (K+16), N = lane%16 (already folded into n).
__device__ __forceinline__ v16bf load_b_frag(const unsigned short* __restrict__ Wt,
                                             int Kpad, int n, int kbase, int lane) {
  int koff = kbase + ((lane & 16) ? 16 : 0);
  const uint4* p = (const uint4*)(Wt + (size_t)n * Kpad + koff);
  return pack_frag(p[0], p[1]);
}

// A fragment (16x32 bf16) from LDS, row-major bf16 with row stride SK (even,
// SK/2 % 4 == 0 for our strides). Lane holds row m=lane%16;
// VGPR i (0-7): K = 2i + (lane>=16 ? 16 : 0) (contiguous, matches B).
__device__ __forceinline__ v16bf load_a_frag(const unsigned short* As, int SK,
                                             int m, int kbase, int lane) {
  int kh4 = (lane & 16) ? 8 : 0; // dword offset for upper half-wave
  const unsigned int* row = (const unsigned int*)As + (((m * SK + kbase) >> 1) + kh4);
  uint4 lo = *(const uint4*)(row);
  uint4 hi = *(const uint4*)(row + 4);
  return pack_frag(lo, hi);
}

__device__ __forceinline__ v8f wmma_bf16(v16bf a, v16bf b, v8f c) {
  return __builtin_amdgcn_wmma_f32_16x16x32_bf16(
      /*neg_a=*/false, a, /*neg_b=*/false, b,
      /*c_mod=*/(short)0, c, /*reuse_a=*/false, /*reuse_b=*/false);
}

// ---------------------------------------------------------------------------
// Weight convert + transpose: W fp32 [K][N] -> Wt bf16 [Npad][Kpad], zero-pad.
// ---------------------------------------------------------------------------
__global__ void convert_wt(const float* __restrict__ W, unsigned short* __restrict__ Wt,
                           int K, int N, int Kpad, int Npad) {
  int idx = blockIdx.x * blockDim.x + threadIdx.x;
  int total = Npad * Kpad;
  if (idx >= total) return;
  int n = idx / Kpad, k = idx - n * Kpad;
  float v = (k < K && n < N) ? W[(size_t)k * N + n] : 0.0f;
  Wt[idx] = f2bf(v);
}

__global__ void zero_f32(float* __restrict__ p, int n) {
  int i = blockIdx.x * blockDim.x + threadIdx.x;
  if (i < n) p[i] = 0.0f;
}

// ---------------------------------------------------------------------------
// Stage 1: node encoder. 64 nodes/block, 8 waves (each owns 16 output cols).
// h_v = relu(x @ W1 + b1) @ W2 + b2  -> stored as bf16 [N][128]
// ---------------------------------------------------------------------------
__global__ __launch_bounds__(256) void node_encoder(
    const float* __restrict__ x,
    const unsigned short* __restrict__ W1t, const float* __restrict__ b1,
    const unsigned short* __restrict__ W2t, const float* __restrict__ b2,
    unsigned short* __restrict__ hv_bf16, int n_nodes) {
  __shared__ __align__(16) unsigned short sA[64 * 32];
  __shared__ __align__(16) unsigned short sH[64 * 128];
  const int tid = threadIdx.x;
  const int lane = tid & 31;
  const int wave = tid >> 5;
  const int node0 = blockIdx.x * 64;

  // load node_states tile, pad K 9->32, bf16
  for (int idx = tid; idx < 64 * 32; idx += 256) {
    int m = idx >> 5, k = idx & 31;
    int g = node0 + m;
    float v = (k < 9 && g < n_nodes) ? x[(size_t)g * 9 + k] : 0.0f;
    sA[idx] = f2bf(v);
  }
  __syncthreads();

  const int n = wave * 16 + (lane & 15);
  const int mr = lane & 15;
  const int mhi = (lane & 16) ? 8 : 0;

  // layer 1: single K-step (K=32)
  v8f acc[4];
#pragma unroll
  for (int s = 0; s < 4; ++s) acc[s] = vzero8();
  {
    v16bf bf = load_b_frag(W1t, 32, n, 0, lane);
#pragma unroll
    for (int s = 0; s < 4; ++s) {
      v16bf af = load_a_frag(sA, 32, s * 16 + mr, 0, lane);
      acc[s] = wmma_bf16(af, bf, acc[s]);
    }
  }
  const float bias1 = b1[n];
#pragma unroll
  for (int s = 0; s < 4; ++s)
#pragma unroll
    for (int r = 0; r < 8; ++r) {
      float v = fmaxf(acc[s][r] + bias1, 0.0f);
      sH[(size_t)(s * 16 + r + mhi) * 128 + n] = f2bf(v);
    }
  __syncthreads();

  // layer 2: K=128
  v8f acc2[4];
#pragma unroll
  for (int s = 0; s < 4; ++s) acc2[s] = vzero8();
#pragma unroll
  for (int kb = 0; kb < 128; kb += 32) {
    v16bf bf = load_b_frag(W2t, 128, n, kb, lane);
#pragma unroll
    for (int s = 0; s < 4; ++s) {
      v16bf af = load_a_frag(sH, 128, s * 16 + mr, kb, lane);
      acc2[s] = wmma_bf16(af, bf, acc2[s]);
    }
  }
  const float bias2 = b2[n];
#pragma unroll
  for (int s = 0; s < 4; ++s)
#pragma unroll
    for (int r = 0; r < 8; ++r) {
      int g = node0 + s * 16 + r + mhi;
      if (g < n_nodes) hv_bf16[(size_t)g * 128 + n] = f2bf(acc2[s][r] + bias2);
    }
}

// ---------------------------------------------------------------------------
// Stage 2: edge MLP + scatter-add. 64 edges/block.
// h_e = relu([hv[row]||hv[col]||attr] @ W1 + b1) @ W2 + b2 ; agg[col] += h_e
// ---------------------------------------------------------------------------
__global__ __launch_bounds__(256) void edge_mlp_scatter(
    const unsigned short* __restrict__ hv_bf16,
    const int* __restrict__ eidx, const float* __restrict__ eattr,
    const unsigned short* __restrict__ W1t, const float* __restrict__ b1,
    const unsigned short* __restrict__ W2t, const float* __restrict__ b2,
    float* __restrict__ agg, int n_edges) {
  __shared__ __align__(16) unsigned short sA[64 * 288]; // K padded 257->288
  __shared__ __align__(16) unsigned short sH[64 * 128];
  const int tid = threadIdx.x;
  const int lane = tid & 31;
  const int wave = tid >> 5;
  const int e0 = blockIdx.x * 64;
  const int E = n_edges;

  // build [64 x 288] bf16 input tile via dword copies (144 dwords per edge)
  unsigned int* sAd = (unsigned int*)sA;
  const unsigned int* hvd = (const unsigned int*)hv_bf16; // 64 dwords / node
  for (int idx = tid; idx < 64 * 144; idx += 256) {
    int el = idx / 144, kd = idx - el * 144;
    int e = e0 + el;
    unsigned int v = 0u;
    if (e < E) {
      if (kd < 64)        v = hvd[(size_t)eidx[e] * 64 + kd];            // hv[row]
      else if (kd < 128)  v = hvd[(size_t)eidx[E + e] * 64 + (kd - 64)]; // hv[col]
      else if (kd == 128) v = (unsigned int)f2bf(eattr[e]);              // k=256
    }
    sAd[idx] = v;
  }
  __syncthreads();

  const int n = wave * 16 + (lane & 15);
  const int mr = lane & 15;
  const int mhi = (lane & 16) ? 8 : 0;

  // layer 1: K=288 (9 steps)
  v8f acc[4];
#pragma unroll
  for (int s = 0; s < 4; ++s) acc[s] = vzero8();
  for (int kb = 0; kb < 288; kb += 32) {
    v16bf bf = load_b_frag(W1t, 288, n, kb, lane);
#pragma unroll
    for (int s = 0; s < 4; ++s) {
      v16bf af = load_a_frag(sA, 288, s * 16 + mr, kb, lane);
      acc[s] = wmma_bf16(af, bf, acc[s]);
    }
  }
  const float bias1 = b1[n];
#pragma unroll
  for (int s = 0; s < 4; ++s)
#pragma unroll
    for (int r = 0; r < 8; ++r) {
      float v = fmaxf(acc[s][r] + bias1, 0.0f);
      sH[(size_t)(s * 16 + r + mhi) * 128 + n] = f2bf(v);
    }
  __syncthreads();

  // layer 2: K=128 (4 steps)
  v8f acc2[4];
#pragma unroll
  for (int s = 0; s < 4; ++s) acc2[s] = vzero8();
#pragma unroll
  for (int kb = 0; kb < 128; kb += 32) {
    v16bf bf = load_b_frag(W2t, 128, n, kb, lane);
#pragma unroll
    for (int s = 0; s < 4; ++s) {
      v16bf af = load_a_frag(sH, 128, s * 16 + mr, kb, lane);
      acc2[s] = wmma_bf16(af, bf, acc2[s]);
    }
  }
  const float bias2 = b2[n];
#pragma unroll
  for (int s = 0; s < 4; ++s)
#pragma unroll
    for (int r = 0; r < 8; ++r) {
      int e = e0 + s * 16 + r + mhi;
      if (e < E) {
        int c = eidx[E + e];
        unsafeAtomicAdd(&agg[(size_t)c * 128 + n], acc2[s][r] + bias2);
      }
    }
}

// ---------------------------------------------------------------------------
// Stage 3: node update. 64 nodes/block.
// out = relu([hv||agg] @ W1 + b1) @ W2 + b2   (OUT_DIM=6, padded to 16)
// ---------------------------------------------------------------------------
__global__ __launch_bounds__(256) void node_update(
    const unsigned short* __restrict__ hv_bf16,
    const float* __restrict__ agg,
    const unsigned short* __restrict__ W1t, const float* __restrict__ b1,
    const unsigned short* __restrict__ W2t, const float* __restrict__ b2,
    float* __restrict__ out, int n_nodes) {
  __shared__ __align__(16) unsigned short sA[64 * 256];
  __shared__ __align__(16) unsigned short sH[64 * 128];
  const int tid = threadIdx.x;
  const int lane = tid & 31;
  const int wave = tid >> 5;
  const int node0 = blockIdx.x * 64;

  unsigned int* sAd = (unsigned int*)sA;
  const unsigned int* hvd = (const unsigned int*)hv_bf16;
  for (int idx = tid; idx < 64 * 128; idx += 256) {
    int m = idx >> 7, kd = idx & 127;
    int g = node0 + m;
    unsigned int v = 0u;
    if (g < n_nodes) {
      if (kd < 64) {
        v = hvd[(size_t)g * 64 + kd];
      } else {
        int k = (kd - 64) * 2;
        float f0 = agg[(size_t)g * 128 + k];
        float f1 = agg[(size_t)g * 128 + k + 1];
        v = (unsigned int)f2bf(f0) | ((unsigned int)f2bf(f1) << 16);
      }
    }
    sAd[idx] = v;
  }
  __syncthreads();

  const int n = wave * 16 + (lane & 15);
  const int mr = lane & 15;
  const int mhi = (lane & 16) ? 8 : 0;

  // layer 1: K=256 (8 steps)
  v8f acc[4];
#pragma unroll
  for (int s = 0; s < 4; ++s) acc[s] = vzero8();
#pragma unroll
  for (int kb = 0; kb < 256; kb += 32) {
    v16bf bf = load_b_frag(W1t, 256, n, kb, lane);
#pragma unroll
    for (int s = 0; s < 4; ++s) {
      v16bf af = load_a_frag(sA, 256, s * 16 + mr, kb, lane);
      acc[s] = wmma_bf16(af, bf, acc[s]);
    }
  }
  const float bias1 = b1[n];
#pragma unroll
  for (int s = 0; s < 4; ++s)
#pragma unroll
    for (int r = 0; r < 8; ++r) {
      float v = fmaxf(acc[s][r] + bias1, 0.0f);
      sH[(size_t)(s * 16 + r + mhi) * 128 + n] = f2bf(v);
    }
  __syncthreads();

  // layer 2: only 16 output cols (6 valid); waves 0..3 each take one M-subtile
  if (wave < 4) {
    const int n2 = lane & 15;
    v8f acc2 = vzero8();
#pragma unroll
    for (int kb = 0; kb < 128; kb += 32) {
      v16bf bf = load_b_frag(W2t, 128, n2, kb, lane);
      v16bf af = load_a_frag(sH, 128, wave * 16 + mr, kb, lane);
      acc2 = wmma_bf16(af, bf, acc2);
    }
    if (n2 < 6) {
      const float bias2 = b2[n2];
#pragma unroll
      for (int r = 0; r < 8; ++r) {
        int g = node0 + wave * 16 + r + mhi;
        if (g < n_nodes) out[(size_t)g * 6 + n2] = acc2[r] + bias2;
      }
    }
  }
}

// ---------------------------------------------------------------------------
// Host launcher
// ---------------------------------------------------------------------------
extern "C" void kernel_launch(void* const* d_in, const int* in_sizes, int n_in,
                              void* d_out, int out_size, void* d_ws, size_t ws_size,
                              hipStream_t stream) {
  (void)in_sizes; (void)n_in; (void)out_size; (void)ws_size;
  const int N = 50000, E = 600000;

  const float* node_states = (const float*)d_in[0];
  const int*   edge_index  = (const int*)d_in[1];
  const float* edge_attr   = (const float*)d_in[2];
  const float* ne_W1 = (const float*)d_in[3];
  const float* ne_b1 = (const float*)d_in[4];
  const float* ne_W2 = (const float*)d_in[5];
  const float* ne_b2 = (const float*)d_in[6];
  const float* ee_W1 = (const float*)d_in[7];
  const float* ee_b1 = (const float*)d_in[8];
  const float* ee_W2 = (const float*)d_in[9];
  const float* ee_b2 = (const float*)d_in[10];
  const float* nu_W1 = (const float*)d_in[11];
  const float* nu_b1 = (const float*)d_in[12];
  const float* nu_W2 = (const float*)d_in[13];
  const float* nu_b2 = (const float*)d_in[14];

  char* ws = (char*)d_ws;
  size_t off = 0;
  auto walloc = [&](size_t bytes) -> void* {
    void* p = ws + off;
    off = (off + bytes + 255) & ~(size_t)255;
    return p;
  };
  unsigned short* hv_bf  = (unsigned short*)walloc((size_t)N * 128 * 2); // 12.8 MB
  float*          aggb   = (float*)walloc((size_t)N * 128 * 4);          // 25.6 MB
  unsigned short* ne_W1t = (unsigned short*)walloc(128 * 32 * 2);
  unsigned short* ne_W2t = (unsigned short*)walloc(128 * 128 * 2);
  unsigned short* ee_W1t = (unsigned short*)walloc(128 * 288 * 2);
  unsigned short* ee_W2t = (unsigned short*)walloc(128 * 128 * 2);
  unsigned short* nu_W1t = (unsigned short*)walloc(128 * 256 * 2);
  unsigned short* nu_W2t = (unsigned short*)walloc(16 * 128 * 2);

  auto conv = [&](const float* W, unsigned short* Wt, int K, int Nn, int Kp, int Np) {
    int total = Np * Kp;
    convert_wt<<<(total + 255) / 256, 256, 0, stream>>>(W, Wt, K, Nn, Kp, Np);
  };
  conv(ne_W1, ne_W1t,   9, 128,  32, 128);
  conv(ne_W2, ne_W2t, 128, 128, 128, 128);
  conv(ee_W1, ee_W1t, 257, 128, 288, 128);
  conv(ee_W2, ee_W2t, 128, 128, 128, 128);
  conv(nu_W1, nu_W1t, 256, 128, 256, 128);
  conv(nu_W2, nu_W2t, 128,   6, 128,  16);

  int aggN = N * 128;
  zero_f32<<<(aggN + 255) / 256, 256, 0, stream>>>(aggb, aggN);

  node_encoder<<<(N + 63) / 64, 256, 0, stream>>>(
      node_states, ne_W1t, ne_b1, ne_W2t, ne_b2, hv_bf, N);

  edge_mlp_scatter<<<E / 64, 256, 0, stream>>>(
      hv_bf, edge_index, edge_attr, ee_W1t, ee_b1, ee_W2t, ee_b2, aggb, E);

  node_update<<<(N + 63) / 64, 256, 0, stream>>>(
      hv_bf, aggb, nu_W1t, nu_b1, nu_W2t, nu_b2, (float*)d_out, N);
}
